// VectorQuantizer_5488968204711
// MI455X (gfx1250) — matrix operand order repaired and checked by reading, not compiled, not run on hardware
//
#include <hip/hip_runtime.h>

typedef __attribute__((ext_vector_type(16))) _Float16 v16h;
typedef __attribute__((ext_vector_type(8)))  float    v8f;
typedef __attribute__((ext_vector_type(4)))  float    f4;

#define NROWS  32768
#define KCODES 8192
#define DDIM   256
#define KTILES (KCODES / 16)   // 512 k-tiles of 16 codes

// ---------------------------------------------------------------------------
// Kernel 1: cn[k] = 0.5 * |c_k|^2   (fp32, exact)
// One wave per code; 8 codes per 256-thread block.
// ---------------------------------------------------------------------------
__global__ void vq_cnorm(const float* __restrict__ cb, float* __restrict__ cn) {
  const int wave = threadIdx.x >> 5;
  const int lane = threadIdx.x & 31;
  const int k    = blockIdx.x * 8 + wave;
  const float* row = cb + (size_t)k * DDIM;
  float s = 0.f;
  for (int i = lane; i < DDIM; i += 32) { float v = row[i]; s += v * v; }
  #pragma unroll
  for (int m = 16; m >= 1; m >>= 1) s += __shfl_xor(s, m, 32);
  if (lane == 0) cn[k] = 0.5f * s;
}

// ---------------------------------------------------------------------------
// Kernel 2: fused split-f16 WMMA GEMM + running argmax.
// Block = 256 threads = 8 waves; each wave owns a 16-row M-tile (128 rows/blk).
// K streamed in 16-code tiles through double-buffered LDS (hi/lo f16).
// score = x.c - 0.5|c|^2 ; argmax(score) == argmin(d2).
// Three independent accumulators break WMMA->WMMA RAW chains; B fragments
// are software-pipelined one chunk ahead to hide LDS latency behind WMMAs.
// ---------------------------------------------------------------------------
__launch_bounds__(256, 1)
__global__ void vq_argmax(const float* __restrict__ x,
                          const float* __restrict__ cb,
                          const float* __restrict__ cn,
                          int* __restrict__ outIdx) {
  // B-fragment-swizzled tiles: [buf][q*512 + lane*16 + j]
  //   value = c[16t + (lane&15)][32q + (lane<16?0:16) + j]
  __shared__ __attribute__((aligned(32))) _Float16 sBh[2][8 * 512];
  __shared__ __attribute__((aligned(32))) _Float16 sBl[2][8 * 512];

  const int tid  = threadIdx.x;
  const int wave = tid >> 5;
  const int lane = tid & 31;
  const int n16  = lane & 15;
  const int hi16 = lane >> 4;          // 0 or 1

  // ---- A fragments (16-bit A 16x32 layout): resident hi/lo f16 in VGPRs ----
  const int rtile = blockIdx.x * 8 + wave;               // 16-row tile
  const float* xrow = x + (size_t)(rtile * 16 + n16) * DDIM;
  v16h Ah[8], Al[8];
  #pragma unroll
  for (int q = 0; q < 8; ++q) {
    const int d0 = 32 * q + 16 * hi16;   // halves j=0..7  -> K = 16*half + j
    const int d1 = d0 + 8;               // halves j=8..15 -> K = 16*half + j
    #pragma unroll
    for (int j = 0; j < 8; ++j) {
      float v0 = xrow[d0 + j];
      float v1 = xrow[d1 + j];
      _Float16 h0 = (_Float16)v0, h1 = (_Float16)v1;
      Ah[q][j]     = h0;  Al[q][j]     = (_Float16)(v0 - (float)h0);
      Ah[q][j + 8] = h1;  Al[q][j + 8] = (_Float16)(v1 - (float)h1);
    }
  }

  // ---- staging role: wave == chunk q, each thread moves 16 contiguous floats
  const int sdbase = 32 * wave + 16 * hi16;

  f4 rr0, rr1, rr2, rr3;
  auto stage_load = [&](int t) {
    const float* p = cb + (size_t)(t * 16 + n16) * DDIM + sdbase;
    rr0 = *(const f4*)(p + 0);  rr1 = *(const f4*)(p + 4);
    rr2 = *(const f4*)(p + 8);  rr3 = *(const f4*)(p + 12);
  };
  auto stage_store = [&](int b) {
    float vals[16];
    *(f4*)&vals[0] = rr0; *(f4*)&vals[4]  = rr1;
    *(f4*)&vals[8] = rr2; *(f4*)&vals[12] = rr3;
    v16h hv, lv;
    #pragma unroll
    for (int j = 0; j < 16; ++j) {
      _Float16 h = (_Float16)vals[j];
      hv[j] = h;
      lv[j] = (_Float16)(vals[j] - (float)h);
    }
    *(v16h*)&sBh[b][wave * 512 + lane * 16] = hv;
    *(v16h*)&sBl[b][wave * 512 + lane * 16] = lv;
  };

  stage_load(0);
  stage_store(0);
  __syncthreads();

  float bestv[8];
  int   bidxv[8];
  #pragma unroll
  for (int j = 0; j < 8; ++j) { bestv[j] = -3.4e38f; bidxv[j] = 0; }

  for (int t = 0; t < KTILES; ++t) {
    const int cur = t & 1;
    if (t + 1 < KTILES) stage_load(t + 1);   // overlap global loads with WMMA

    v8f C0 = {}, C1 = {}, C2 = {};           // independent split-product chains
    // B fragments pipelined one chunk ahead of their consuming WMMAs
    v16h Bh = *(const v16h*)&sBh[cur][lane * 16];
    v16h Bl = *(const v16h*)&sBl[cur][lane * 16];
    #pragma unroll
    for (int q = 0; q < 8; ++q) {
      v16h curBh = Bh, curBl = Bl;
      if (q < 7) {
        Bh = *(const v16h*)&sBh[cur][(q + 1) * 512 + lane * 16];
        Bl = *(const v16h*)&sBl[cur][(q + 1) * 512 + lane * 16];
      }
      C0 = __builtin_amdgcn_wmma_f32_16x16x32_f16(false, Ah[q], false, curBh, (short)0, C0, false, false);
      C1 = __builtin_amdgcn_wmma_f32_16x16x32_f16(false, Ah[q], false, curBl, (short)0, C1, false, false);
      C2 = __builtin_amdgcn_wmma_f32_16x16x32_f16(false, Al[q], false, curBh, (short)0, C2, false, false);
    }

    const float cnv = cn[t * 16 + n16];      // 0.5|c|^2 for this lane's code
    const int   myk = t * 16 + n16;
    #pragma unroll
    for (int j = 0; j < 8; ++j) {
      float s = (C0[j] + C1[j] + C2[j]) - cnv;
      if (s > bestv[j]) { bestv[j] = s; bidxv[j] = myk; }   // '>' keeps earliest k
    }

    if (t + 1 < KTILES) stage_store(cur ^ 1);
    __syncthreads();
  }

  // ---- cross-lane argmax over the 16-lane N extent of each C row ----
  #pragma unroll
  for (int j = 0; j < 8; ++j) {
    float b = bestv[j]; int bi = bidxv[j];
    #pragma unroll
    for (int m = 1; m < 16; m <<= 1) {       // xor<16 stays within each half
      float ob = __shfl_xor(b, m, 32);
      int   oi = __shfl_xor(bi, m, 32);
      if (ob > b || (ob == b && oi < bi)) { b = ob; bi = oi; }
    }
    if (lane == j)      outIdx[rtile * 16 + j]     = bi;   // rows 0..7
    if (lane == 16 + j) outIdx[rtile * 16 + 8 + j] = bi;   // rows 8..15
  }
}

// ---------------------------------------------------------------------------
// Kernel 3: gather + output assembly.  out = [z_q | z | x | idx(float)]
// z_q == z numerically (straight-through estimator).
// ---------------------------------------------------------------------------
__global__ void vq_gather(const float* __restrict__ x,
                          const float* __restrict__ cb,
                          const int* __restrict__ idx,
                          float* __restrict__ out) {
  const size_t ND  = (size_t)NROWS * DDIM;
  const int    row = blockIdx.x;
  const int    d   = threadIdx.x;
  const int    k   = idx[row];
  const float  c   = cb[(size_t)k * DDIM + d];
  const size_t o   = (size_t)row * DDIM + d;
  out[o]            = c;            // z_q
  out[ND + o]       = c;            // z
  out[2 * ND + o]   = x[o];         // x passthrough
  if (d == 0) out[3 * ND + row] = (float)k;   // indices (exact in f32)
}

// ---------------------------------------------------------------------------
extern "C" void kernel_launch(void* const* d_in, const int* in_sizes, int n_in,
                              void* d_out, int out_size, void* d_ws, size_t ws_size,
                              hipStream_t stream) {
  (void)in_sizes; (void)n_in; (void)out_size; (void)ws_size;
  const float* x  = (const float*)d_in[0];
  const float* cb = (const float*)d_in[1];

  float* cn  = (float*)d_ws;                                   // 32 KB
  int*   idx = (int*)((char*)d_ws + KCODES * sizeof(float));   // 128 KB

  vq_cnorm <<<KCODES / 8,  256, 0, stream>>>(cb, cn);
  vq_argmax<<<NROWS / 128, 256, 0, stream>>>(x, cb, cn, idx);
  vq_gather<<<NROWS,       256, 0, stream>>>(x, cb, idx, (float*)d_out);
}
